// RelRepNoContext_9388798509772
// MI455X (gfx1250) — matrix operand order repaired and checked by reading, not compiled, not run on hardware
//
#include <hip/hip_runtime.h>

// ---------------------------------------------------------------------------
// RelRepNoContext fused kernel for gfx1250 (MI455X, wave32, WMMA + TDM):
//   out[b,r,:] = relu(concat(span[b,head],span[b,tail]) @ W1 + b1) @ W2 + b2
// B=4, S=1024, H=256, R=65536, FFN=768.  bf16 WMMA, f32 accumulation.
//
// Pipeline:
//   prep kernel : W1 (512x768 f32) -> W1T (768x512 bf16), W2 (768x256 f32)
//                 -> W2T (256x768 bf16) in workspace (1.13 MB, L2-resident).
//   main kernel : per 128-row tile, gather X -> LDS (bf16); per 64-wide FFN
//                 chunk, stage W1T/W2T slices into padded LDS tiles with the
//                 Tensor Data Mover (tensor_load_to_lds, TENSORcnt), then
//                 GEMM1 -> relu -> GEMM2 entirely in v_wmma_f32_16x16x32_bf16.
// ---------------------------------------------------------------------------

typedef __attribute__((ext_vector_type(16))) __bf16 v16bf;
typedef __attribute__((ext_vector_type(8)))  __bf16 v8bf;
typedef __attribute__((ext_vector_type(4)))  __bf16 v4bf;
typedef __attribute__((ext_vector_type(8)))  float  v8f;
typedef __attribute__((ext_vector_type(4)))  float  v4f;
typedef __attribute__((ext_vector_type(4)))  unsigned int v4u;
typedef __attribute__((ext_vector_type(8)))  int    v8i;
typedef __attribute__((ext_vector_type(4)))  int    v4i;

namespace cfg {
constexpr int Hd   = 256;    // hidden
constexpr int Sd   = 1024;   // spans
constexpr int Rd   = 65536;  // relations per batch (power of two)
constexpr int FFNd = 768;
constexpr int K1   = 512;    // 2*H
constexpr int MT   = 128;    // rows per workgroup (8 waves x 16 rows)
constexpr int FC   = 64;     // FFN chunk per LDS stage

// LDS leading dims (bf16 elements), padded (+8 elems = +16B = +4 DWORDs)
constexpr int XLD  = K1 + 8;   // 520
constexpr int W1LD = K1 + 8;   // 520
constexpr int W2LD = FC + 8;   // 72
constexpr int HLD  = FC + 8;   // 72

constexpr int XT_ELEMS  = MT  * XLD;    // 66560
constexpr int W1T_ELEMS = FC  * W1LD;   // 33280
constexpr int W2T_ELEMS = Hd  * W2LD;   // 18432
constexpr int HT_ELEMS  = 8 * 16 * HLD; // 9216
constexpr int SMEM_BYTES =
    (XT_ELEMS + W1T_ELEMS + W2T_ELEMS + HT_ELEMS) * (int)sizeof(__bf16); // 254976

// workspace layout (bf16 elements)
constexpr int W1TG_ELEMS = FFNd * K1;   // 393216  (row n, col k)
constexpr int W2TG_ELEMS = Hd * FFNd;   // 196608  (row n, col k)
} // namespace cfg

// ---------------------------------------------------------------------------
// TDM 2-D padded load: global (bf16, row stride stride0 elems) -> LDS tile
// with pad_amount DWORD padding inserted every pad_interval DWORDs.
// Descriptor bit layout per CDNA5 ISA ch.10.8 (D# groups 0/1; groups 2/3 = 0
// for 2-D tiles). All operands are wave-uniform -> land in SGPRs.
// ---------------------------------------------------------------------------
static __device__ __forceinline__ void tdm_load_2d(
    unsigned int lds_byte_addr, const void* gaddr,
    unsigned int tile_d0, unsigned int tile_d1,
    unsigned long long stride0_elems,
    unsigned int pad_interval_code, unsigned int pad_amount_code)
{
  const unsigned long long ga = (unsigned long long)(uintptr_t)gaddr;
  v4u g0;
  g0[0] = 1u;                                        // count=1, user descriptor
  g0[1] = lds_byte_addr;                             // lds_addr [63:32]
  g0[2] = (unsigned int)(ga & 0xffffffffu);          // global_addr[31:0]  -> [95:64]
  g0[3] = (unsigned int)((ga >> 32) & 0x01ffffffu)   // global_addr[56:32] -> [120:96]
        | (2u << 30);                                // type=2 ("image")   -> [127:126]

  const unsigned int flags =
      (1u << 16)                    // data_size = 1 -> 2 bytes (bf16)
    | (1u << 20)                    // pad_enable (LDS destination padding)
    | (pad_interval_code << 22)     // [24:22]
    | (pad_amount_code  << 25);     // [31:25]
  // tensor dims == tile dims (tile fully in-bounds; no clipping wanted)
  const unsigned int td0 = tile_d0, td1 = tile_d1;
  v8i g1;
  g1[0] = (int)flags;
  g1[1] = (int)((td0 & 0xffffu) << 16);                                  // dim0[15:0] -> [63:48]
  g1[2] = (int)(((td0 >> 16) & 0xffffu) | ((td1 & 0xffffu) << 16));      // dim0[31:16], dim1[15:0]
  g1[3] = (int)(((td1 >> 16) & 0xffffu) | ((tile_d0 & 0xffffu) << 16));  // dim1[31:16], tile_dim0
  g1[4] = (int)(tile_d1 & 0xffffu);                                      // tile_dim1 (tile_dim2=0)
  g1[5] = (int)(unsigned int)(stride0_elems & 0xffffffffu);              // dim0_stride[31:0]
  g1[6] = (int)(unsigned int)((stride0_elems >> 32) & 0xffffu);          // dim0_stride[47:32]
  g1[7] = 0;                                                             // dim1_stride unused (2-D)

  const v4i z4 = {0, 0, 0, 0};
#if defined(__clang_major__) && (__clang_major__ >= 23)
  const v8i z8 = {0, 0, 0, 0, 0, 0, 0, 0};
  __builtin_amdgcn_tensor_load_to_lds(g0, g1, z4, z4, z8, 0);
#else
  __builtin_amdgcn_tensor_load_to_lds(g0, g1, z4, z4, 0);
#endif
}

// Build a v16bf lane-fragment from two 16-byte LDS chunks (off8 elems apart).
static __device__ __forceinline__ v16bf frag16(const __bf16* p, int off8) {
  v8bf lo = *(const v8bf*)(p);
  v8bf hi = *(const v8bf*)(p + off8);
  return __builtin_shufflevector(lo, hi, 0, 1, 2, 3, 4, 5, 6, 7,
                                         8, 9, 10, 11, 12, 13, 14, 15);
}

static __device__ __forceinline__ v8f wmma_bf16(v16bf a, v16bf b, v8f c) {
  // 8 args: (neg_a, A, neg_b, B, c_mod, C, reuse_a, reuse_b)
  return __builtin_amdgcn_wmma_f32_16x16x32_bf16(false, a, false, b,
                                                 (short)0, c, false, false);
}

// ---------------------------------------------------------------------------
// Prep: transpose + convert weights to bf16 once per call (1.13 MB -> L2).
//   w1t[n*512 + k] = bf16(W1[k*768 + n])   n<768, k<512
//   w2t[n*768 + k] = bf16(W2[k*256 + n])   n<256, k<768
// ---------------------------------------------------------------------------
__global__ __launch_bounds__(256)
void prep_weights_kernel(const float* __restrict__ W1, const float* __restrict__ W2,
                         __bf16* __restrict__ w1t, __bf16* __restrict__ w2t) {
  using namespace cfg;
  const int idx = blockIdx.x * 256 + threadIdx.x;
  if (idx < W1TG_ELEMS) {
    const int n = idx >> 9;          // /512
    const int k = idx & (K1 - 1);
    w1t[idx] = (__bf16)W1[(size_t)k * FFNd + n];
  } else {
    const int j = idx - W1TG_ELEMS;
    if (j < W2TG_ELEMS) {
      const int n = j / FFNd;
      const int k = j - n * FFNd;
      w2t[j] = (__bf16)W2[(size_t)k * Hd + n];
    }
  }
}

// ---------------------------------------------------------------------------
// Main fused kernel.
// ---------------------------------------------------------------------------
__global__ __launch_bounds__(256)
void relrep_fused_kernel(const float* __restrict__ span,      // (B,S,H) f32
                         const long long* __restrict__ rel,   // (B,R,2) i64
                         const __bf16* __restrict__ w1t,      // (FFN,512) bf16
                         const float* __restrict__ b1,        // (FFN)
                         const __bf16* __restrict__ w2t,      // (H,FFN) bf16
                         const float* __restrict__ b2,        // (H)
                         float* __restrict__ out)             // (B,R,H) f32
{
  using namespace cfg;
  extern __shared__ __align__(16) char smem_raw[];
  __bf16* XT   = (__bf16*)smem_raw;          // [MT][XLD]   gathered X, bf16
  __bf16* W1T  = XT  + XT_ELEMS;             // [FC][W1LD]  W1 chunk (n,k), TDM-filled
  __bf16* W2T  = W1T + W1T_ELEMS;            // [Hd][W2LD]  W2 chunk (n,k), TDM-filled
  __bf16* HT   = W2T + W2T_ELEMS;            // [8][16][HLD] per-wave hidden patch

  const int tid  = threadIdx.x;
  const int wave = tid >> 5;        // 0..7 (wave32)
  const int lane = tid & 31;
  const int l0   = lane & 15;       // fragment row/col id
  const int hi   = lane >> 4;       // lane half select

  // -------------------------------------------------------------------------
  // Stage 0: gather 128 rows of concat(head,tail) -> LDS (f32 -> bf16)
  // -------------------------------------------------------------------------
  {
    const int row  = tid >> 1;                 // 0..127
    const int half = tid & 1;                  // 0 = head, 1 = tail
    const long long m  = (long long)blockIdx.x * MT + row;   // global row
    const int bb = (int)(m >> 16);             // m / R
    const int rr = (int)(m & (Rd - 1));        // m % R
    const long long sidx = rel[(((long long)bb * Rd + rr) << 1) + half];
    const float* src = span + ((size_t)bb * Sd + (size_t)sidx) * Hd;
    __bf16* dst = XT + row * XLD + half * Hd;
#pragma unroll 8
    for (int j = 0; j < Hd; j += 8) {
      v4f f0 = *(const v4f*)(src + j);
      v4f f1 = *(const v4f*)(src + j + 4);
      v4bf g0, g1;
      g0[0] = (__bf16)f0[0]; g0[1] = (__bf16)f0[1];
      g0[2] = (__bf16)f0[2]; g0[3] = (__bf16)f0[3];
      g1[0] = (__bf16)f1[0]; g1[1] = (__bf16)f1[1];
      g1[2] = (__bf16)f1[2]; g1[3] = (__bf16)f1[3];
      *(v4bf*)(dst + j)     = g0;
      *(v4bf*)(dst + j + 4) = g1;
    }
  }

  const v8f vzero = {0.f, 0.f, 0.f, 0.f, 0.f, 0.f, 0.f, 0.f};
  v8f oacc[16];                       // 16x256 f32 output tile per wave
#pragma unroll
  for (int i = 0; i < 16; ++i) oacc[i] = vzero;

  const __bf16* xbase = XT + (wave * 16 + l0) * XLD;   // A rows for this wave
  __bf16* hpatch = HT + wave * 16 * HLD;
  const unsigned int w1t_lds = (unsigned int)(uintptr_t)W1T;  // LDS byte offset
  const unsigned int w2t_lds = (unsigned int)(uintptr_t)W2T;

  // -------------------------------------------------------------------------
  // FFN chunk loop: TDM-stage W1/W2 slices, GEMM1 -> relu -> GEMM2.
  // -------------------------------------------------------------------------
  for (int fc = 0; fc < FFNd; fc += FC) {
    __syncthreads();   // previous chunk's W1T/W2T reads complete

    if (wave == 0) {
      // W1T chunk: rows fc..fc+63 of (768x512) -> LDS [64][520]
      //   row = 512 elems = 1024 B = 256 DW; pad 4 DW every 256 DW
      tdm_load_2d(w1t_lds, w1t + (size_t)fc * K1,
                  /*tile_d0=*/K1, /*tile_d1=*/FC, /*stride0=*/K1,
                  /*pad_interval: 7 -> 256 DW*/ 7u, /*pad_amount: 3 -> 4 DW*/ 3u);
      // W2T chunk: cols fc..fc+63 of (256x768) -> LDS [256][72]
      //   row = 64 elems = 128 B = 32 DW; pad 4 DW every 32 DW
      tdm_load_2d(w2t_lds, w2t + fc,
                  /*tile_d0=*/FC, /*tile_d1=*/Hd, /*stride0=*/FFNd,
                  /*pad_interval: 4 -> 32 DW*/ 4u, /*pad_amount: 3 -> 4 DW*/ 3u);
      __builtin_amdgcn_s_wait_tensorcnt(0);
    }
    __syncthreads();   // TDM data visible to all waves

    // --- GEMM1: hacc(16x64) = X(16x512) @ W1chunk(512x64) ---
    v8f hacc[4];
#pragma unroll
    for (int i = 0; i < 4; ++i) hacc[i] = vzero;
#pragma unroll 4
    for (int kt = 0; kt < K1 / 32; ++kt) {
      // A frag: halves at K = kt*32 + 8*hi and +16
      v16bf a = frag16(xbase + kt * 32 + 8 * hi, 16);
#pragma unroll
      for (int nt = 0; nt < 4; ++nt) {
        // B frag: halves at K = kt*32 + 8*hi and +16, row N = nt*16 + l0
        v16bf b = frag16(W1T + (nt * 16 + l0) * W1LD + kt * 32 + 8 * hi, 16);
        hacc[nt] = wmma_bf16(a, b, hacc[nt]);
      }
    }

    // --- bias + ReLU, round to bf16, re-layout via per-wave LDS patch ---
#pragma unroll
    for (int nt = 0; nt < 4; ++nt) {
      const float bias = b1[fc + nt * 16 + l0];
#pragma unroll
      for (int v = 0; v < 8; ++v) {
        float x = hacc[nt][v] + bias;            // C-frag: M = v + 8*hi, N = l0
        x = x > 0.f ? x : 0.f;
        hpatch[(v + 8 * hi) * HLD + nt * 16 + l0] = (__bf16)x;
      }
    }
    // same-wave LDS RAW: DScnt ordering inserted by compiler

    // --- GEMM2: oacc(16x256) += h(16x64) @ W2chunk(64x256) ---
    const __bf16* habase = hpatch + l0 * HLD;
#pragma unroll
    for (int kt = 0; kt < FC / 32; ++kt) {
      v16bf a = frag16(habase + kt * 32 + 8 * hi, 16);
#pragma unroll
      for (int nt = 0; nt < 16; ++nt) {
        v16bf b = frag16(W2T + (nt * 16 + l0) * W2LD + kt * 32 + 8 * hi, 16);
        oacc[nt] = wmma_bf16(a, b, oacc[nt]);
      }
    }
  }

  // -------------------------------------------------------------------------
  // Epilogue: + b2, store f32. C-frag element (v): row M = v+8*hi, col = nt*16+l0
  // -------------------------------------------------------------------------
  const long long rowbase = (long long)blockIdx.x * MT + wave * 16;
#pragma unroll
  for (int nt = 0; nt < 16; ++nt) {
    const float bias = b2[nt * 16 + l0];
#pragma unroll
    for (int v = 0; v < 8; ++v) {
      const size_t row = (size_t)(rowbase + v + 8 * hi);
      out[row * Hd + nt * 16 + l0] = oacc[nt][v] + bias;
    }
  }
}

extern "C" void kernel_launch(void* const* d_in, const int* in_sizes, int n_in,
                              void* d_out, int out_size, void* d_ws, size_t ws_size,
                              hipStream_t stream) {
  (void)in_sizes; (void)n_in; (void)out_size; (void)ws_size;
  using namespace cfg;
  const float*     span = (const float*)d_in[0];      // (B,S,H)   f32
  const long long* rel  = (const long long*)d_in[1];  // (B,R,2)   i64
  const float*     W1   = (const float*)d_in[2];      // (2H,FFN)  f32
  const float*     b1   = (const float*)d_in[3];      // (FFN)
  const float*     W2   = (const float*)d_in[4];      // (FFN,H)   f32
  const float*     b2   = (const float*)d_in[5];      // (H)
  float*           out  = (float*)d_out;              // (B,R,H)   f32

  // workspace: bf16 transposed weights, 1,179,648 bytes total
  __bf16* w1t = (__bf16*)d_ws;                        // 768 x 512
  __bf16* w2t = w1t + W1TG_ELEMS;                     // 256 x 768

  // prep: convert + transpose weights (runs every call; deterministic)
  {
    const int total = W1TG_ELEMS + W2TG_ELEMS;        // 589824
    prep_weights_kernel<<<(total + 255) / 256, 256, 0, stream>>>(W1, W2, w1t, w2t);
  }

  // 249 KB dynamic LDS per workgroup (within 320 KB WGP budget); opt in.
  (void)hipFuncSetAttribute((const void*)relrep_fused_kernel,
                            hipFuncAttributeMaxDynamicSharedMemorySize,
                            SMEM_BYTES);

  const int total_rows = 4 * Rd;                 // B * R = 262144
  const int grid = total_rows / MT;              // 2048 workgroups
  relrep_fused_kernel<<<grid, 256, SMEM_BYTES, stream>>>(
      span, rel, w1t, b1, w2t, b2, out);
}
